// CrossAttention_24129126269276
// MI455X (gfx1250) — matrix operand order repaired and checked by reading, not compiled, not run on hardware
//
#include <hip/hip_runtime.h>

// ---------------------------------------------------------------------------
// CDNA5 (gfx1250) cross-attention, f16 WMMA path, 64x64 wave tiles,
// async global->LDS staging (GLOBAL_LOAD_ASYNC_TO_LDS_B128 / ASYNCcnt).
// ---------------------------------------------------------------------------

typedef _Float16 v16h __attribute__((ext_vector_type(16)));
typedef _Float16 v8h  __attribute__((ext_vector_type(8)));
typedef float    v8f  __attribute__((ext_vector_type(8)));
typedef float    v4f  __attribute__((ext_vector_type(4)));
typedef int      v4i  __attribute__((ext_vector_type(4)));

static constexpr int kB    = 16;
static constexpr int kHW   = 4096;   // 64*64
static constexpr int kC    = 640;
static constexpr int kCTX  = 768;
static constexpr int kH    = 8;
static constexpr int kD    = 80;     // head dim
static constexpr int kT    = 77;     // context tokens
static constexpr int kTP   = 96;     // padded tokens / padded head K-dim

#define WMMA_F16(a, b, c) \
  __builtin_amdgcn_wmma_f32_16x16x32_f16(false, (a), false, (b), (short)0, (c), false, false)

#ifndef __has_builtin
#define __has_builtin(x) 0
#endif
#if __has_builtin(__builtin_amdgcn_global_load_async_to_lds_b128)
#define HAVE_ASYNC_LDS 1
#else
#define HAVE_ASYNC_LDS 0
#endif

typedef __attribute__((address_space(1))) v4i* as1_v4i_p;
typedef __attribute__((address_space(3))) v4i* as3_v4i_p;

// 16-byte global -> LDS copy: async DMA (ASYNCcnt) when available, else via VGPR.
static __device__ inline void copy16_g2l(const _Float16* g, _Float16* l) {
#if HAVE_ASYNC_LDS
  __builtin_amdgcn_global_load_async_to_lds_b128((as1_v4i_p)g, (as3_v4i_p)l, 0, 0);
#else
  *(v8h*)l = *(const v8h*)g;
#endif
}

static __device__ inline void copy_wait() {
#if HAVE_ASYNC_LDS
#if __has_builtin(__builtin_amdgcn_s_wait_asynccnt)
  __builtin_amdgcn_s_wait_asynccnt(0);
#else
  asm volatile("s_wait_asynccnt 0x0" ::: "memory");
#endif
#endif
}

// Build a 16-half fragment from two aligned 8-half LDS loads.
static __device__ inline v16h frag_from2(const _Float16* p0, const _Float16* p1) {
  const v8h x = *(const v8h*)p0;
  const v8h y = *(const v8h*)p1;
  return __builtin_shufflevector(x, y, 0, 1, 2, 3, 4, 5, 6, 7,
                                 8, 9, 10, 11, 12, 13, 14, 15);
}

// A-matrix 16x32 (MxK) from row-major LDS tile [m][k], pitch in halves.
static __device__ inline v16h frag_a(const _Float16* s, int pitch) {
  const int lane = threadIdx.x & 31;
  const int l = lane & 15, hi = lane >> 4;
  const _Float16* r = s + l * pitch + hi * 8;
  return frag_from2(r, r + 16);
}

// B-matrix 32x16 (KxN) from *transposed* LDS tile [n][k], pitch in halves.
static __device__ inline v16h frag_bt(const _Float16* s, int pitch) {
  const int lane = threadIdx.x & 31;
  const int l = lane & 15, hi = lane >> 4;
  const _Float16* r = s + l * pitch + hi * 8;
  return frag_from2(r, r + 16);
}

// --- conversion / transpose kernels -----------------------------------------

__global__ void k_cvt_f32_f16(const float* __restrict__ src,
                              _Float16* __restrict__ dst, int n) {
  int i = blockIdx.x * 256 + threadIdx.x;
  if (i < n) dst[i] = (_Float16)src[i];
}

__global__ void k_zero_f16(_Float16* __restrict__ dst, int n) {
  int i = blockIdx.x * 256 + threadIdx.x;
  if (i < n) dst[i] = (_Float16)0.f;
}

// dst[n][k] = (f16) src[k][n] ; K, N multiples of 32. block (32,8).
__global__ void k_transpose_w(const float* __restrict__ src,
                              _Float16* __restrict__ dst, int K, int N) {
  __shared__ _Float16 t[32][33];
  const int n0 = blockIdx.x * 32, k0 = blockIdx.y * 32;
  const int tx = threadIdx.x, ty = threadIdx.y;
#pragma unroll
  for (int j = 0; j < 32; j += 8)
    t[ty + j][tx] = (_Float16)src[(long)(k0 + ty + j) * N + n0 + tx];
  __syncthreads();
#pragma unroll
  for (int j = 0; j < 32; j += 8)
    dst[(long)(n0 + ty + j) * K + k0 + tx] = t[tx][ty + j];
}

// Xt[b][hw][c] = (f16) x[b][c][hw] ; block (32,8), grid (HW/32, C/32, B).
__global__ void k_transpose_x(const float* __restrict__ x,
                              _Float16* __restrict__ Xt) {
  __shared__ _Float16 t[32][33];
  const int hw0 = blockIdx.x * 32, c0 = blockIdx.y * 32, b = blockIdx.z;
  const int tx = threadIdx.x, ty = threadIdx.y;
  const float* src = x + ((long)b * kC + c0) * kHW + hw0;
#pragma unroll
  for (int j = 0; j < 32; j += 8)
    t[ty + j][tx] = (_Float16)src[(long)(ty + j) * kHW + tx];  // t[c][hw]
  __syncthreads();
  _Float16* dst = Xt + ((long)b * kHW + hw0) * kC + c0;
#pragma unroll
  for (int j = 0; j < 32; j += 8)
    dst[(long)(ty + j) * kC + tx] = t[tx][ty + j];
}

// --- batched GEMM: Out[b] = A[b] (MxK, f16 row-major) @ Wt^T + bias ---------
// Wt: transposed weights, [n][k] row-major, ld = K.
// MODE 0: f16 row-major at Out + b*bO + m*ldO + n
// MODE 1: f32 transposed at Out + b*bO + n*ldO + m   (final projection)
// Block tile 256x128, BK=32, 8 waves (4 M x 2 N), wave tile 64x64.
template <int MODE, bool CHECK_M>
__global__ __launch_bounds__(256) void k_gemm(
    const _Float16* __restrict__ A, long sAm, long bA, int M, int K,
    const _Float16* __restrict__ Wt,
    const float* __restrict__ bias,
    void* __restrict__ Out, long ldO, long bO) {
  constexpr int AP = 40;   // 32 + 8 pad halves (80B rows: 16B aligned)
  constexpr int BP = 40;
  __shared__ __align__(16) _Float16 As[256 * AP];   // [m][k]  20.0 KB
  __shared__ __align__(16) _Float16 Bs[128 * BP];   // [n][k]  10.0 KB

  const int tid = threadIdx.x;
  const int bidx = blockIdx.z;
  const int m0 = blockIdx.y * 256;
  const int n0 = blockIdx.x * 128;
  const _Float16* Ab = A + (long)bidx * bA;

  const int wave = tid >> 5, lane = tid & 31, l = lane & 15, hi = lane >> 4;
  const int wm = (wave >> 1) * 64;  // 4 waves along M
  const int wn = (wave & 1) * 64;   // 2 waves along N

  v8f acc[4][4] = {};

  for (int k0 = 0; k0 < K; k0 += 32) {
    // stage A (256x32): 16B packets, async DMA to LDS when available
#pragma unroll
    for (int p = 0; p < 4; ++p) {
      const int idx = tid + p * 256;            // 1024 packets
      const int m = idx >> 2, kk8 = (idx & 3) * 8;
      const int gm = m0 + m;
      _Float16* dst = As + m * AP + kk8;
      if (!CHECK_M || gm < M) {
        copy16_g2l(Ab + (long)gm * sAm + (k0 + kk8), dst);
      } else {
        v8h z = {};
        *(v8h*)dst = z;
      }
    }
    // stage Wt (128x32)
#pragma unroll
    for (int p = 0; p < 2; ++p) {
      const int idx = tid + p * 256;            // 512 packets
      const int n = idx >> 2, kk8 = (idx & 3) * 8;
      copy16_g2l(Wt + (long)(n0 + n) * K + (k0 + kk8), Bs + n * BP + kk8);
    }
    // prefetch next K-tile (gfx1250 global_prefetch_b8)
    if (k0 + 32 < K) {
      __builtin_prefetch(Wt + (long)(n0 + (tid & 127)) * K + (k0 + 32), 0, 1);
      __builtin_prefetch(Ab + (long)(m0 + tid) * sAm + (k0 + 32), 0, 1);
    }
    copy_wait();
    __syncthreads();

    v16h a[4];
#pragma unroll
    for (int i = 0; i < 4; ++i) a[i] = frag_a(&As[(wm + i * 16) * AP], AP);
#pragma unroll
    for (int j = 0; j < 4; ++j) {
      const v16h bj = frag_bt(&Bs[(wn + j * 16) * BP], BP);
#pragma unroll
      for (int i = 0; i < 4; ++i) acc[i][j] = WMMA_F16(a[i], bj, acc[i][j]);
    }
    __syncthreads();
  }

  _Float16* Oh = (_Float16*)Out;
  float* Of = (float*)Out;
#pragma unroll
  for (int i = 0; i < 4; ++i)
#pragma unroll
    for (int j = 0; j < 4; ++j) {
      const int n = n0 + wn + j * 16 + l;
      const float bn = bias[n];
      if (MODE == 0) {
#pragma unroll
        for (int v = 0; v < 8; ++v) {
          const int m = m0 + wm + i * 16 + v + 8 * hi;
          if (!CHECK_M || m < M)
            Oh[(long)bidx * bO + (long)m * ldO + n] = (_Float16)(acc[i][j][v] + bn);
        }
      } else {
        // per lane, v spans 8 consecutive m -> two b128 stores
        const int mb = m0 + wm + i * 16 + 8 * hi;
        v4f lo, hv;
#pragma unroll
        for (int u = 0; u < 4; ++u) {
          lo[u] = acc[i][j][u] + bn;
          hv[u] = acc[i][j][4 + u] + bn;
        }
        float* dst = Of + (long)bidx * bO + (long)n * ldO + mb;
        *(v4f*)dst = lo;
        *(v4f*)(dst + 4) = hv;
      }
    }
}

// --- fused attention: softmax(Q K^T * scale) V, per (128-query tile, head, b)
__global__ __launch_bounds__(256) void k_attn(
    const _Float16* __restrict__ Q,   // (B, HW, C)
    const _Float16* __restrict__ Kt,  // (B, 96, C)  rows 77..95 are zero
    const _Float16* __restrict__ Vt,  // (B, 96, C)  rows 77..95 are zero
    _Float16* __restrict__ S) {       // (B, HW, C)
  constexpr int QP = 104;  // 96 + 8 pad halves (208B: 16B aligned)
  constexpr int KP = 104;
  constexpr int VP = 104;
  __shared__ __align__(16) _Float16 Qs[128 * QP];  // [row][dim]; reused for P [row][t]
  __shared__ __align__(16) _Float16 Ks[kTP * KP];  // [t][dim]  (B of scores)
  __shared__ __align__(16) _Float16 Vs[kD * VP];   // [dim][t]  (B of PV)

  const int tid = threadIdx.x;
  const int row0 = blockIdx.x * 128;
  const int head = blockIdx.y;
  const int b = blockIdx.z;

  const long qbase = ((long)b * kHW + row0) * kC + head * kD;
  const long kbase = ((long)b * kTP) * kC + head * kD;

  // stage Q tile: 2 threads per row, 6 packets each (dims 80..95 zero)
  {
    const int r = tid >> 1, half = tid & 1;
#pragma unroll
    for (int j = 0; j < 6; ++j) {
      const int kk8 = half * 48 + j * 8;
      _Float16* dst = Qs + r * QP + kk8;
      if (kk8 < kD) {
        copy16_g2l(Q + qbase + (long)r * kC + kk8, dst);
      } else {
        v8h z = {};
        *(v8h*)dst = z;
      }
    }
  }
  // stage K as [t][dim] and V transposed [dim][t]
  if (tid < 192) {
    const int t = tid >> 1, half = tid & 1;
#pragma unroll
    for (int j = 0; j < 6; ++j) {
      const int kk8 = half * 48 + j * 8;
      _Float16* dst = Ks + t * KP + kk8;
      if (kk8 < kD) {
        copy16_g2l(Kt + kbase + (long)t * kC + kk8, dst);
      } else {
        v8h z = {};
        *(v8h*)dst = z;
      }
    }
#pragma unroll
    for (int j = 0; j < 5; ++j) {
      const int kk8 = half * 40 + j * 8;   // 0..72, all < 80
      const v8h vv = *(const v8h*)(Vt + kbase + (long)t * kC + kk8);
#pragma unroll
      for (int u = 0; u < 8; ++u) Vs[(kk8 + u) * VP + t] = vv[u];
    }
  }
  copy_wait();
  __syncthreads();

  const int wave = tid >> 5, lane = tid & 31, l = lane & 15, hi = lane >> 4;
  const int wm = wave * 16;  // each of 8 waves owns 16 query rows

  // scores: (16 x 96) @ (96 x 80) -> 5 N-tiles, 3 K-steps
  v8f acc[5] = {};
#pragma unroll
  for (int ks = 0; ks < 3; ++ks) {
    const v16h a = frag_a(&Qs[wm * QP + ks * 32], QP);
#pragma unroll
    for (int nt = 0; nt < 5; ++nt) {
      const v16h bb = frag_bt(&Ks[(nt * 16) * KP + ks * 32], KP);
      acc[nt] = WMMA_F16(a, bb, acc[nt]);
    }
  }

  // scale + mask + softmax over T (row M lives on 16 lanes of a half-wave)
  const float scale = 0.111803398874989485f;  // 80^-0.5
#pragma unroll
  for (int v = 0; v < 8; ++v) {
    float mx = -3.0e38f;
#pragma unroll
    for (int nt = 0; nt < 5; ++nt) {
      const int t = nt * 16 + l;
      const float x = (t < kT) ? acc[nt][v] * scale : -3.0e38f;
      acc[nt][v] = x;
      mx = fmaxf(mx, x);
    }
#pragma unroll
    for (int mk = 1; mk < 16; mk <<= 1) mx = fmaxf(mx, __shfl_xor(mx, mk, 32));
    float sum = 0.f;
#pragma unroll
    for (int nt = 0; nt < 5; ++nt) {
      const float e = __expf(acc[nt][v] - mx);
      acc[nt][v] = e;
      sum += e;
    }
#pragma unroll
    for (int mk = 1; mk < 16; mk <<= 1) sum += __shfl_xor(sum, mk, 32);
    const float inv = 1.0f / sum;
    const int r = wm + v + 8 * hi;  // own rows -> no cross-wave hazard
#pragma unroll
    for (int nt = 0; nt < 5; ++nt) {
      const int t = nt * 16 + l;
      Qs[r * QP + t] = (_Float16)(acc[nt][v] * inv);
    }
    Qs[r * QP + 80 + l] = (_Float16)0.f;  // zero K-dim padding t=80..95
  }
  __syncthreads();

  // out: (16 x 96) @ (96 x 80) ; B from Vs [n=dim][k=t]
  v8f acc2[5] = {};
#pragma unroll
  for (int ks = 0; ks < 3; ++ks) {
    const v16h a = frag_a(&Qs[wm * QP + ks * 32], QP);
#pragma unroll
    for (int nt = 0; nt < 5; ++nt) {
      const v16h bb = frag_bt(&Vs[(nt * 16) * VP + ks * 32], VP);
      acc2[nt] = WMMA_F16(a, bb, acc2[nt]);
    }
  }

  const long sbase = ((long)b * kHW + row0) * kC + head * kD;
#pragma unroll
  for (int nt = 0; nt < 5; ++nt)
#pragma unroll
    for (int v = 0; v < 8; ++v) {
      const int m = wm + v + 8 * hi;
      const int n = nt * 16 + l;
      S[sbase + (long)m * kC + n] = (_Float16)acc2[nt][v];
    }
}

// ---------------------------------------------------------------------------

extern "C" void kernel_launch(void* const* d_in, const int* in_sizes, int n_in,
                              void* d_out, int out_size, void* d_ws, size_t ws_size,
                              hipStream_t stream) {
  (void)in_sizes; (void)n_in; (void)out_size; (void)ws_size;

  const float* x   = (const float*)d_in[0];  // (B, C, H, W)
  const float* ctx = (const float*)d_in[1];  // (B, T, CTX)
  const float* Wq  = (const float*)d_in[2];
  const float* bq  = (const float*)d_in[3];
  const float* Wk  = (const float*)d_in[4];
  const float* bk  = (const float*)d_in[5];
  const float* Wv  = (const float*)d_in[6];
  const float* bv  = (const float*)d_in[7];
  const float* Wo  = (const float*)d_in[8];
  const float* bo  = (const float*)d_in[9];
  float* out = (float*)d_out;

  char* base = (char*)d_ws;
  size_t off = 0;
  auto alloc = [&](size_t elems) -> _Float16* {
    _Float16* p = (_Float16*)(base + off);
    off += (elems * sizeof(_Float16) + 255) & ~(size_t)255;
    return p;
  };
  _Float16* WqT  = alloc((size_t)kC * kC);           // [n][k]
  _Float16* WkT  = alloc((size_t)kC * kCTX);
  _Float16* WvT  = alloc((size_t)kC * kCTX);
  _Float16* WoT  = alloc((size_t)kC * kC);
  _Float16* ctxh = alloc((size_t)kB * kT * kCTX);
  _Float16* Xt   = alloc((size_t)kB * kHW * kC);     // x transposed, f16
  _Float16* Qh   = alloc((size_t)kB * kHW * kC);
  _Float16* Kh   = alloc((size_t)kB * kTP * kC);
  _Float16* Vh   = alloc((size_t)kB * kTP * kC);
  _Float16* Sh   = alloc((size_t)kB * kHW * kC);

  const dim3 tb(32, 8);
  k_transpose_w<<<dim3(kC / 32, kC / 32), tb, 0, stream>>>(Wq, WqT, kC, kC);
  k_transpose_w<<<dim3(kC / 32, kCTX / 32), tb, 0, stream>>>(Wk, WkT, kCTX, kC);
  k_transpose_w<<<dim3(kC / 32, kCTX / 32), tb, 0, stream>>>(Wv, WvT, kCTX, kC);
  k_transpose_w<<<dim3(kC / 32, kC / 32), tb, 0, stream>>>(Wo, WoT, kC, kC);
  k_cvt_f32_f16<<<(kB * kT * kCTX + 255) / 256, 256, 0, stream>>>(
      ctx, ctxh, kB * kT * kCTX);
  k_transpose_x<<<dim3(kHW / 32, kC / 32, kB), tb, 0, stream>>>(x, Xt);

  k_zero_f16<<<(kB * kTP * kC + 255) / 256, 256, 0, stream>>>(Kh, kB * kTP * kC);
  k_zero_f16<<<(kB * kTP * kC + 255) / 256, 256, 0, stream>>>(Vh, kB * kTP * kC);

  // Q = Xt @ Wq + bq
  k_gemm<0, false><<<dim3(kC / 128, kHW / 256, kB), 256, 0, stream>>>(
      Xt, (long)kC, (long)kHW * kC, kHW, kC,
      WqT, bq, Qh, (long)kC, (long)kHW * kC);

  // K = ctx @ Wk + bk  (M=77 -> zero-padded 96-row buffer)
  k_gemm<0, true><<<dim3(kC / 128, 1, kB), 256, 0, stream>>>(
      ctxh, (long)kCTX, (long)kT * kCTX, kT, kCTX,
      WkT, bk, Kh, (long)kC, (long)kTP * kC);

  // V = ctx @ Wv + bv
  k_gemm<0, true><<<dim3(kC / 128, 1, kB), 256, 0, stream>>>(
      ctxh, (long)kCTX, (long)kT * kCTX, kT, kCTX,
      WvT, bv, Vh, (long)kC, (long)kTP * kC);

  // fused attention -> Sh (B, HW, C)
  k_attn<<<dim3(kHW / 128, kH, kB), 256, 0, stream>>>(Qh, Kh, Vh, Sh);

  // out = (S @ Wo + bo)^T  -> f32 (B, C, HW)
  k_gemm<1, false><<<dim3(kC / 128, kHW / 256, kB), 256, 0, stream>>>(
      Sh, (long)kC, (long)kHW * kC, kHW, kC,
      WoT, bo, out, (long)kHW, (long)kC * kHW);
}
